// SparseMoE_29188597743839
// MI455X (gfx1250) — matrix operand order repaired and checked by reading, not compiled, not run on hardware
//
#include <hip/hip_runtime.h>
#include <stdint.h>

#define MTOK 8192   // B*S tokens
#define DDIM 1024   // embed
#define HDIM 4096   // ffn hidden
#define NEXP 8

typedef __attribute__((ext_vector_type(16))) __bf16 v16bf;
typedef __attribute__((ext_vector_type(8)))  float  v8f;

__device__ __forceinline__ unsigned short f2bf(float f) {
  unsigned int u = __float_as_uint(f);
  unsigned int r = (u + 0x7FFFu + ((u >> 16) & 1u)) >> 16;  // round-nearest-even
  return (unsigned short)r;
}

union FragBF {
  v16bf v;
  uint4 q[2];
};

// CDNA5 async global->LDS DMA (ASYNCcnt). vdst = LDS byte address (low 32 bits
// of flat pointer within the LDS aperture), vaddr = 64-bit global address.
__device__ __forceinline__ void async_load_b128(unsigned lds_addr, const void* gptr) {
  asm volatile("global_load_async_to_lds_b128 %0, %1, off"
               :: "v"(lds_addr), "v"(gptr) : "memory");
}
__device__ __forceinline__ void wait_async0() {
  asm volatile("s_wait_asynccnt 0" ::: "memory");
}
__device__ __forceinline__ unsigned lds_off(const void* p) {
  return (unsigned)(size_t)p;  // LDS aperture: addr[31:0] == LDS byte offset
}

// ---------------------------------------------------------------------------
// Router: one wave per token. gates[n] = renormalized top-2 softmax values
// (collapses to a 2-way softmax over the top-2 logits).
// ---------------------------------------------------------------------------
__global__ __launch_bounds__(256) void router_kernel(
    const float* __restrict__ x, const float* __restrict__ Wr,
    const float* __restrict__ br, float* __restrict__ gates) {
  const int lane = threadIdx.x & 31;
  const int wave = threadIdx.x >> 5;
  const int n = blockIdx.x * 8 + wave;
  const float* xr = x + (size_t)n * DDIM;

  float acc[NEXP];
#pragma unroll
  for (int e = 0; e < NEXP; ++e) acc[e] = 0.f;

  for (int d = lane; d < DDIM; d += 32) {
    const float xv = xr[d];
    const float4* w = (const float4*)(Wr + (size_t)d * NEXP);
    const float4 w0 = w[0], w1 = w[1];
    acc[0] += xv * w0.x; acc[1] += xv * w0.y;
    acc[2] += xv * w0.z; acc[3] += xv * w0.w;
    acc[4] += xv * w1.x; acc[5] += xv * w1.y;
    acc[6] += xv * w1.z; acc[7] += xv * w1.w;
  }
#pragma unroll
  for (int off = 16; off > 0; off >>= 1)
#pragma unroll
    for (int e = 0; e < NEXP; ++e) acc[e] += __shfl_xor(acc[e], off, 32);

  if (lane == 0) {
    float m1 = -1e30f, m2 = -1e30f;
#pragma unroll
    for (int e = 0; e < NEXP; ++e) {
      const float l = acc[e] + br[e];
      if (l > m1) { m2 = m1; m1 = l; } else if (l > m2) { m2 = l; }
    }
    const float g0 = 1.f / (1.f + __expf(m2 - m1));
    gates[2 * n + 0] = g0;
    gates[2 * n + 1] = 1.f - g0;
  }
}

// ---------------------------------------------------------------------------
// fp32 -> bf16 elementwise (4 elems / thread)
// ---------------------------------------------------------------------------
__global__ __launch_bounds__(256) void cvt_bf16_kernel(
    const float* __restrict__ in, unsigned short* __restrict__ out) {
  const size_t i = (size_t)blockIdx.x * blockDim.x + threadIdx.x;
  const float4 v = ((const float4*)in)[i];
  ushort4 o;
  o.x = f2bf(v.x); o.y = f2bf(v.y); o.z = f2bf(v.z); o.w = f2bf(v.w);
  ((ushort4*)out)[i] = o;
}

// ---------------------------------------------------------------------------
// Tiled transpose + convert: in fp32 [z][R][C] -> out bf16 [z][C][R]
// ---------------------------------------------------------------------------
__global__ __launch_bounds__(256) void transpose_cvt_kernel(
    const float* __restrict__ in, unsigned short* __restrict__ out,
    int R, int C) {
  __shared__ float tile[32][33];
  const size_t es = (size_t)R * C;
  in  += es * blockIdx.z;
  out += es * blockIdx.z;
  const int c0 = blockIdx.x * 32, r0 = blockIdx.y * 32;
#pragma unroll
  for (int i = threadIdx.y; i < 32; i += 8)
    tile[i][threadIdx.x] = in[(size_t)(r0 + i) * C + c0 + threadIdx.x];
  __syncthreads();
#pragma unroll
  for (int i = threadIdx.y; i < 32; i += 8)
    out[(size_t)(c0 + i) * R + r0 + threadIdx.x] = f2bf(tile[threadIdx.x][i]);
}

// ---------------------------------------------------------------------------
// bf16 WMMA GEMM: C = A[M][Kd] * Bt[N][Kd]^T (+ bias / relu / gate)
//   MODE 0: Hout[idx] = bf16(relu(acc + bias[col]))
//   MODE 1: Fout[idx]  = gate[row]*(acc + bias[col])
//   MODE 2: Fout[idx] += gate[row]*(acc + bias[col])
// Block: 256 threads (8 waves), 128x128 tile; waves 4x2, wave tile 32x64
// (2x4 wmma tiles). Double-buffered LDS fed by async global->LDS DMA.
// ---------------------------------------------------------------------------
template <int MODE>
__global__ __launch_bounds__(256) void gemm_bf16_wmma(
    const unsigned short* __restrict__ A,   // [M][Kd] bf16
    const unsigned short* __restrict__ Bt,  // [N][Kd] bf16 (pre-transposed)
    const float* __restrict__ bias,         // [N]
    const float* __restrict__ gates,        // [M][2]
    int gk,
    unsigned short* __restrict__ Hout,      // MODE 0
    float* __restrict__ Fout,               // MODE 1/2
    int Kd, int Ndim) {
  // 40-ushort rows: 80B stride keeps 16B alignment and staggers banks.
  __shared__ __align__(16) unsigned short As[2][128][40];
  __shared__ __align__(16) unsigned short Bs[2][128][40];

  const int tid  = threadIdx.x;
  const int lane = tid & 31;
  const int wave = tid >> 5;
  const int wm = wave >> 1;            // 0..3 (M dir)
  const int wn = wave & 1;             // 0..1 (N dir)
  const int hl  = lane >> 4;           // half-wave: selects K-group
  const int l16 = lane & 15;

  const int M0 = blockIdx.y * 128;
  const int N0 = blockIdx.x * 128;

  // Tile copy: 128x32 bf16 = 512 16B-chunks per matrix; 2 chunks/thread each.
  const int r0 = tid >> 2;             // 0..63
  const int cc = (tid & 3) * 8;        // 0,8,16,24 (bf16 elems)
  const int r1 = r0 + 64;

  const unsigned short* Ag0 = A  + (size_t)(M0 + r0) * Kd + cc;
  const unsigned short* Ag1 = A  + (size_t)(M0 + r1) * Kd + cc;
  const unsigned short* Bg0 = Bt + (size_t)(N0 + r0) * Kd + cc;
  const unsigned short* Bg1 = Bt + (size_t)(N0 + r1) * Kd + cc;

  unsigned la0[2], la1[2], lb0[2], lb1[2];
#pragma unroll
  for (int b = 0; b < 2; ++b) {
    la0[b] = lds_off(&As[b][r0][cc]);
    la1[b] = lds_off(&As[b][r1][cc]);
    lb0[b] = lds_off(&Bs[b][r0][cc]);
    lb1[b] = lds_off(&Bs[b][r1][cc]);
  }

  // Prologue: DMA first K-slab into buffer 0.
  async_load_b128(la0[0], Ag0);
  async_load_b128(la1[0], Ag1);
  async_load_b128(lb0[0], Bg0);
  async_load_b128(lb1[0], Bg1);
  wait_async0();
  __syncthreads();

  v8f acc[2][4] = {};
  int cur = 0;
  for (int k0 = 0; k0 < Kd; k0 += 32) {
    const int nxt = cur ^ 1;
    if (k0 + 32 < Kd) {  // prefetch next slab into the other buffer
      async_load_b128(la0[nxt], Ag0 + k0 + 32);
      async_load_b128(la1[nxt], Ag1 + k0 + 32);
      async_load_b128(lb0[nxt], Bg0 + k0 + 32);
      async_load_b128(lb1[nxt], Bg1 + k0 + 32);
    }

    FragBF fa[2], fb[4];
#pragma unroll
    for (int i = 0; i < 2; ++i) {
      // A frag (16x32 bf16): lane = row l16, K in {hl*8..+7, 16+hl*8..+7}
      const int r = wm * 32 + i * 16 + l16;
      fa[i].q[0] = *(const uint4*)&As[cur][r][hl * 8];
      fa[i].q[1] = *(const uint4*)&As[cur][r][16 + hl * 8];
    }
#pragma unroll
    for (int j = 0; j < 4; ++j) {
      // B frag (32x16 bf16): lane = col l16, K in {hl*8..+7, 16+hl*8..+7}
      const int c = wn * 64 + j * 16 + l16;
      fb[j].q[0] = *(const uint4*)&Bs[cur][c][hl * 8];
      fb[j].q[1] = *(const uint4*)&Bs[cur][c][16 + hl * 8];
    }
#pragma unroll
    for (int i = 0; i < 2; ++i)
#pragma unroll
      for (int j = 0; j < 4; ++j)
        acc[i][j] = __builtin_amdgcn_wmma_f32_16x16x32_bf16(
            false, fa[i].v, false, fb[j].v, (short)0, acc[i][j], false, false);

    wait_async0();       // our DMA into buf[nxt] landed
    __syncthreads();     // everyone's landed; all reads of buf[cur] done
    cur = nxt;
  }

  // Epilogue. C/D layout: lane l16 = col, VGPR r -> row (hl*8 + r).
#pragma unroll
  for (int i = 0; i < 2; ++i) {
    const int rbase = M0 + wm * 32 + i * 16 + hl * 8;
#pragma unroll
    for (int j = 0; j < 4; ++j) {
      const int col = N0 + wn * 64 + j * 16 + l16;
      const float bv = bias[col];
#pragma unroll
      for (int r = 0; r < 8; ++r) {
        const int row = rbase + r;
        const float v = acc[i][j][r] + bv;
        const size_t idx = (size_t)row * Ndim + col;
        if constexpr (MODE == 0) {
          Hout[idx] = f2bf(fmaxf(v, 0.f));
        } else {
          const float g = gates[2 * row + gk];
          float o = g * v;
          if constexpr (MODE == 2) o += Fout[idx];
          Fout[idx] = o;
        }
      }
    }
  }
}

// ---------------------------------------------------------------------------
extern "C" void kernel_launch(void* const* d_in, const int* in_sizes, int n_in,
                              void* d_out, int out_size, void* d_ws, size_t ws_size,
                              hipStream_t stream) {
  const float* x  = (const float*)d_in[0];   // [4,2048,1024]
  const float* Wr = (const float*)d_in[1];   // [1024,8]
  const float* br = (const float*)d_in[2];   // [8]
  const float* W1 = (const float*)d_in[3];   // [8,1024,4096]
  const float* b1 = (const float*)d_in[4];   // [8,4096]
  const float* W2 = (const float*)d_in[5];   // [8,4096,1024]
  const float* b2 = (const float*)d_in[6];   // [8,1024]
  float* out = (float*)d_out;                // [4,2048,1024]

  // Workspace layout (~112 MB)
  char* ws = (char*)d_ws;
  unsigned short* xb = (unsigned short*)ws;  ws += (size_t)MTOK * DDIM * 2;      // x bf16
  unsigned short* w1t = (unsigned short*)ws; ws += (size_t)2 * HDIM * DDIM * 2;  // W1[:2]^T bf16 [k][H][D]
  unsigned short* w2t = (unsigned short*)ws; ws += (size_t)2 * DDIM * HDIM * 2;  // W2[:2]^T bf16 [k][D][H]
  float* gates = (float*)ws;                 ws += (size_t)MTOK * 2 * 4;         // [M][2]
  unsigned short* hb = (unsigned short*)ws;  // h bf16 [M][H], reused per expert

  // 1) Router gates
  router_kernel<<<MTOK / 8, 256, 0, stream>>>(x, Wr, br, gates);

  // 2) Precision conversion / pre-transposition
  cvt_bf16_kernel<<<(MTOK * DDIM / 4) / 256, 256, 0, stream>>>(x, xb);
  transpose_cvt_kernel<<<dim3(HDIM / 32, DDIM / 32, 2), dim3(32, 8), 0, stream>>>(
      W1, w1t, DDIM, HDIM);   // [D][H] -> [H][D]
  transpose_cvt_kernel<<<dim3(DDIM / 32, HDIM / 32, 2), dim3(32, 8), 0, stream>>>(
      W2, w2t, HDIM, DDIM);   // [H][D] -> [D][H]

  // 3) Expert chains (stream-ordered; hb reused, out accumulated on k=1)
  for (int k = 0; k < 2; ++k) {
    gemm_bf16_wmma<0><<<dim3(HDIM / 128, MTOK / 128), 256, 0, stream>>>(
        xb, w1t + (size_t)k * HDIM * DDIM, b1 + (size_t)k * HDIM,
        nullptr, 0, hb, nullptr, DDIM, HDIM);
    if (k == 0) {
      gemm_bf16_wmma<1><<<dim3(DDIM / 128, MTOK / 128), 256, 0, stream>>>(
          hb, w2t + (size_t)k * DDIM * HDIM, b2 + (size_t)k * DDIM,
          gates, k, nullptr, out, HDIM, DDIM);
    } else {
      gemm_bf16_wmma<2><<<dim3(DDIM / 128, MTOK / 128), 256, 0, stream>>>(
          hb, w2t + (size_t)k * DDIM * HDIM, b2 + (size_t)k * DDIM,
          gates, k, nullptr, out, HDIM, DDIM);
    }
  }
}
